// RAMTransformer_72404558676324
// MI455X (gfx1250) — 8-bit matrix operand order repaired and checked by reading, compiled, not run on hardware
//
#include <hip/hip_runtime.h>
#include <stdint.h>


typedef int v8i __attribute__((ext_vector_type(8)));

#define RB 2048
#define RK 14
#define RM 16384
#define T1DIM 4096
#define T23DIM 3072
#define N1 2048
#define N23 1024
#define TWORDS 512

__global__ void pack_table_kernel(const float* __restrict__ mem,
                                  unsigned int* __restrict__ out, int nwords) {
    int idx = blockIdx.x * blockDim.x + threadIdx.x;
    if (idx >= nwords) return;
    const float4* p = (const float4*)mem + (size_t)idx * 8;
    unsigned int w = 0;
#pragma unroll
    for (int q = 0; q < 8; ++q) {
        float4 v = p[q];
        w |= (v.x == 1.0f ? 1u : 0u) << (q * 4 + 0);
        w |= (v.y == 1.0f ? 1u : 0u) << (q * 4 + 1);
        w |= (v.z == 1.0f ? 1u : 0u) << (q * 4 + 2);
        w |= (v.w == 1.0f ? 1u : 0u) << (q * 4 + 3);
    }
    out[idx] = w;
}

__global__ void cvt_i32_u8_kernel(const int* __restrict__ src,
                                  unsigned char* __restrict__ dst,
                                  int rowLen, int dstPitch, int dstColOff,
                                  int total4) {
    int idx = blockIdx.x * blockDim.x + threadIdx.x;
    if (idx >= total4) return;
    int r4 = rowLen >> 2;
    int b = idx / r4;
    int c = (idx - b * r4) << 2;
    int4 v = *(const int4*)(src + (size_t)b * rowLen + c);
    uchar4 o = make_uchar4((unsigned char)v.x, (unsigned char)v.y,
                           (unsigned char)v.z, (unsigned char)v.w);
    *(uchar4*)(dst + (size_t)b * dstPitch + dstColOff + c) = o;
}

__global__ void build_w_kernel(const int* __restrict__ conn,
                               unsigned char* __restrict__ wlo,
                               unsigned char* __restrict__ whi,
                               int N, int Kt) {
    int n = blockIdx.x * blockDim.x + threadIdx.x;
    if (n >= N) return;
    int t[RK], w[RK], cnt = 0;
#pragma unroll
    for (int j = 0; j < RK; ++j) {
        int tt = conn[n * RK + j];
        int ww = 1 << (RK - 1 - j);
        bool merged = false;
        for (int i = 0; i < cnt; ++i)
            if (t[i] == tt) { w[i] += ww; merged = true; break; }
        if (!merged) { t[cnt] = tt; w[cnt] = ww; ++cnt; }
    }
    for (int i = 0; i < cnt; ++i) {
        size_t o = (size_t)n * Kt + t[i];
        wlo[o] = (unsigned char)(w[i] & 0xff);
        whi[o] = (unsigned char)((w[i] >> 8) & 0xff);
    }
}

__global__ void __launch_bounds__(256)
ram_gemm_iu8(const unsigned char* __restrict__ A,
             const unsigned char* __restrict__ Wlo,
             const unsigned char* __restrict__ Whi,
             const unsigned int* __restrict__ tbl,
             int Kt, int nTiles32, int mode,
             unsigned char* dst0, unsigned char* dst1,
             int dstPitch, int dstColOff,
             float* outF, int outPitch) {
    const int lane = threadIdx.x & 31;
    const int waveId = blockIdx.x * (blockDim.x >> 5) + (threadIdx.x >> 5);
    const int tN = waveId % nTiles32;
    const int tM = waveId / nTiles32;
    const int m0 = tM * 32, n0 = tN * 32;
    const int lr = lane & 15;
    const int hh = (lane >> 4) & 1;

    const unsigned char* arow0 = A + (size_t)(m0 + lr) * Kt + (hh ? 16 : 0);
    const unsigned char* arow1 = arow0 + (size_t)16 * Kt;
    const size_t brow0 = (size_t)(n0 + lr) * Kt + (hh ? 16 : 0);
    const size_t brow1 = brow0 + (size_t)16 * Kt;

    v8i acc[2][2][2] = {};

    for (int kb = 0; kb < Kt; kb += 64) {
        v8i a[2], bl[2], bh[2];
#pragma unroll
        for (int i = 0; i < 2; ++i) {
            const unsigned char* p = (i ? arow1 : arow0) + kb;
            uint2 c0 = *(const uint2*)(p + 0);
            uint2 c1 = *(const uint2*)(p + 8);
            uint2 c2 = *(const uint2*)(p + 32);
            uint2 c3 = *(const uint2*)(p + 40);
            v8i av = {(int)c0.x, (int)c0.y, (int)c1.x, (int)c1.y,
                      (int)c2.x, (int)c2.y, (int)c3.x, (int)c3.y};
            a[i] = av;
        }
#pragma unroll
        for (int j = 0; j < 2; ++j) {
            size_t base = (j ? brow1 : brow0) + kb;
            int4 l0 = *(const int4*)(Wlo + base);
            int4 l1 = *(const int4*)(Wlo + base + 32);
            v8i blv = {l0.x, l0.y, l0.z, l0.w, l1.x, l1.y, l1.z, l1.w};
            bl[j] = blv;
            int4 h0 = *(const int4*)(Whi + base);
            int4 h1 = *(const int4*)(Whi + base + 32);
            v8i bhv = {h0.x, h0.y, h0.z, h0.w, h1.x, h1.y, h1.z, h1.w};
            bh[j] = bhv;
        }
#pragma unroll
        for (int i = 0; i < 2; ++i)
#pragma unroll
            for (int j = 0; j < 2; ++j) {
                acc[i][j][0] = __builtin_amdgcn_wmma_i32_16x16x64_iu8(
                    false, a[i], false, bl[j], acc[i][j][0], false, false);
                acc[i][j][1] = __builtin_amdgcn_wmma_i32_16x16x64_iu8(
                    false, a[i], false, bh[j], acc[i][j][1], false, false);
            }
    }

#pragma unroll
    for (int i = 0; i < 2; ++i)
#pragma unroll
        for (int j = 0; j < 2; ++j) {
            int n = n0 + j * 16 + lr;
            const unsigned int* trow = tbl + (size_t)n * TWORDS;
#pragma unroll
            for (int v = 0; v < 8; ++v) {
                int m = m0 + i * 16 + hh * 8 + v;
                int addr = acc[i][j][0][v] + (acc[i][j][1][v] << 8);
                unsigned int word = trow[addr >> 5];
                unsigned int bit = (word >> (addr & 31)) & 1u;
                if (mode == 0) {
                    dst0[(size_t)m * dstPitch + n] = (unsigned char)bit;
                    dst1[(size_t)m * dstPitch + n] = (unsigned char)bit;
                } else if (mode == 1) {
                    dst1[(size_t)m * dstPitch + dstColOff + n] = (unsigned char)bit;
                } else {
                    outF[(size_t)m * outPitch + n] = bit ? 1.0f : 0.0f;
                }
            }
        }
}

__global__ void fb_layer_kernel(const int* __restrict__ bi32,
                                const unsigned char* __restrict__ bu8,
                                int srcPitch,
                                const int* __restrict__ conn,
                                const float* __restrict__ mem, int N,
                                unsigned char* dst0, unsigned char* dst1,
                                int dstPitch, int dstColOff,
                                float* outF, int total) {
    int idx = blockIdx.x * blockDim.x + threadIdx.x;
    if (idx >= total) return;
    int b = idx / N, n = idx - b * N;
    const int* cp = conn + n * RK;
    int addr = 0;
#pragma unroll
    for (int j = 0; j < RK; ++j) {
        int c = cp[j];
        int bit = bi32 ? (bi32[(size_t)b * srcPitch + c] & 1)
                       : (int)bu8[(size_t)b * srcPitch + c];
        addr = (addr << 1) | bit;
    }
    float val = mem[(size_t)n * RM + addr];
    unsigned char h = (val == 1.0f) ? 1 : 0;
    if (outF) {
        outF[(size_t)b * N + n] = h ? 1.0f : 0.0f;
    } else {
        dst0[(size_t)b * dstPitch + dstColOff + n] = h;
        if (dst1) dst1[(size_t)b * dstPitch + dstColOff + n] = h;
    }
}

extern "C" void kernel_launch(void* const* d_in, const int* in_sizes, int n_in,
                              void* d_out, int out_size, void* d_ws, size_t ws_size,
                              hipStream_t stream) {
    const int* input_bits = (const int*)d_in[0];
    const int* state_bits = (const int*)d_in[1];
    const int* in_conn    = (const int*)d_in[2];
    const float* in_mem   = (const float*)d_in[3];
    const int* st_conn    = (const int*)d_in[4];
    const float* st_mem   = (const float*)d_in[5];
    const int* out_conn   = (const int*)d_in[6];
    const float* out_mem  = (const float*)d_in[7];
    float* out = (float*)d_out;

    uint8_t* base = (uint8_t*)d_ws;
    size_t off = 0;
    auto take = [&](size_t s) -> void* {
        void* p = base + off;
        off += (s + 255) & ~(size_t)255;
        return p;
    };

    const size_t SZ_W1  = (size_t)N1 * T1DIM;
    const size_t SZ_W23 = (size_t)N23 * T23DIM;
    const size_t SZ_A1  = (size_t)RB * T1DIM;
    const size_t SZ_A23 = (size_t)RB * T23DIM;
    const size_t SZ_T1  = (size_t)N1 * TWORDS * 4;
    const size_t SZ_T23 = (size_t)N23 * TWORDS * 4;

    uint8_t* Wlo1 = (uint8_t*)take(SZ_W1);
    uint8_t* Whi1 = (uint8_t*)take(SZ_W1);
    uint8_t* Wlo2 = (uint8_t*)take(SZ_W23);
    uint8_t* Whi2 = (uint8_t*)take(SZ_W23);
    uint8_t* Wlo3 = (uint8_t*)take(SZ_W23);
    uint8_t* Whi3 = (uint8_t*)take(SZ_W23);
    size_t wBytes = off;
    uint8_t* A1 = (uint8_t*)take(SZ_A1);
    uint8_t* A2 = (uint8_t*)take(SZ_A23);
    uint8_t* A3 = (uint8_t*)take(SZ_A23);
    unsigned int* T1 = (unsigned int*)take(SZ_T1);
    unsigned int* T2 = (unsigned int*)take(SZ_T23);
    unsigned int* T3 = (unsigned int*)take(SZ_T23);
    size_t need = off;

    if (ws_size >= need) {
        hipMemsetAsync(base, 0, wBytes, stream);

        build_w_kernel<<<(N1 + 255) / 256, 256, 0, stream>>>(in_conn, Wlo1, Whi1, N1, T1DIM);
        build_w_kernel<<<(N23 + 255) / 256, 256, 0, stream>>>(st_conn, Wlo2, Whi2, N23, T23DIM);
        build_w_kernel<<<(N23 + 255) / 256, 256, 0, stream>>>(out_conn, Wlo3, Whi3, N23, T23DIM);

        pack_table_kernel<<<(N1 * TWORDS + 255) / 256, 256, 0, stream>>>(in_mem, T1, N1 * TWORDS);
        pack_table_kernel<<<(N23 * TWORDS + 255) / 256, 256, 0, stream>>>(st_mem, T2, N23 * TWORDS);
        pack_table_kernel<<<(N23 * TWORDS + 255) / 256, 256, 0, stream>>>(out_mem, T3, N23 * TWORDS);

        int tot1 = RB * T1DIM / 4;
        cvt_i32_u8_kernel<<<(tot1 + 255) / 256, 256, 0, stream>>>(
            input_bits, A1, T1DIM, T1DIM, 0, tot1);
        int tots = RB * N23 / 4;
        cvt_i32_u8_kernel<<<(tots + 255) / 256, 256, 0, stream>>>(
            state_bits, A2, N23, T23DIM, 2048, tots);

        int waves1 = (RB / 32) * (N1 / 32);
        ram_gemm_iu8<<<waves1 / 8, 256, 0, stream>>>(
            A1, Wlo1, Whi1, T1, T1DIM, N1 / 32,  0,
            A2, A3, T23DIM, 0, nullptr, 0);

        int waves2 = (RB / 32) * (N23 / 32);
        ram_gemm_iu8<<<waves2 / 8, 256, 0, stream>>>(
            A2, Wlo2, Whi2, T2, T23DIM, N23 / 32,  1,
            nullptr, A3, T23DIM, 2048, nullptr, 0);

        ram_gemm_iu8<<<waves2 / 8, 256, 0, stream>>>(
            A3, Wlo3, Whi3, T3, T23DIM, N23 / 32,  2,
            nullptr, nullptr, 0, 0, out, N23);
    } else if (ws_size >= 2 * SZ_A23 + 512) {
        uint8_t* A2f = base;
        uint8_t* A3f = base + ((SZ_A23 + 255) & ~(size_t)255);

        int tots = RB * N23 / 4;
        cvt_i32_u8_kernel<<<(tots + 255) / 256, 256, 0, stream>>>(
            state_bits, A2f, N23, T23DIM, 2048, tots);

        int totA = RB * N1;
        fb_layer_kernel<<<(totA + 255) / 256, 256, 0, stream>>>(
            input_bits, nullptr, T1DIM, in_conn, in_mem, N1,
            A2f, A3f, T23DIM, 0, nullptr, totA);

        int totB = RB * N23;
        fb_layer_kernel<<<(totB + 255) / 256, 256, 0, stream>>>(
            nullptr, A2f, T23DIM, st_conn, st_mem, N23,
            A3f, nullptr, T23DIM, 2048, nullptr, totB);

        fb_layer_kernel<<<(totB + 255) / 256, 256, 0, stream>>>(
            nullptr, A3f, T23DIM, out_conn, out_mem, N23,
            nullptr, nullptr, 0, 0, out, totB);
    }
}
